// Conv_Block_9852654977029
// MI455X (gfx1250) — matrix operand order repaired and checked by reading, not compiled, not run on hardware
//
#include <hip/hip_runtime.h>
#include <hip/hip_bf16.h>

// ---------------- problem constants ----------------
#define NB   32
#define CIN  256
#define HH   56
#define WW   56
#define COUT 256
#define OH   56
#define OWD  58
#define MTOT (NB*OH*OWD)        // 103936 (= 64 * 1624 exactly)
#define KTOT (9*CIN)            // 2304
#define HW   (HH*WW)            // 3136

// ---------------- tiling ----------------
#define COUT_TILE 128
#define PIX_TILE  64
#define KSTAGE    128                   // half a tap: tap constant within a stage
#define NSTAGE    (KTOT/KSTAGE)         // 18
#define APITCH    136                   // padded LDS pitch (halves): 272B -> bank spread
#define BPITCH    136

typedef __attribute__((ext_vector_type(16))) _Float16 v16h;
typedef __attribute__((ext_vector_type(8)))  _Float16 v8h;
typedef __attribute__((ext_vector_type(4)))  _Float16 v4h;
typedef __attribute__((ext_vector_type(8)))  float    v8f;
typedef __attribute__((ext_vector_type(4)))  int      v4i;

union Frag16 { v16h v; v8h h[2]; };

#if defined(__has_builtin)
#if __has_builtin(__builtin_amdgcn_global_load_async_to_lds_b128)
#define HAVE_ASYNC_LDS 1
#endif
#endif

#ifdef HAVE_ASYNC_LDS
typedef __attribute__((address_space(1))) v4i* gas_v4i_p;   // global int4*
typedef __attribute__((address_space(3))) v4i* las_v4i_p;   // LDS    int4*
#endif

// ---------------------------------------------------------------
// 0) zero the per-channel accumulators
// ---------------------------------------------------------------
__global__ void zero_stats_kernel(float* gSum, float* gSq) {
    int t = threadIdx.x;
    if (t < COUT) gSum[t] = 0.0f;
    else          gSq[t - COUT] = 0.0f;
}

// ---------------------------------------------------------------
// 1) binarize weights to f16 {+1,-1}; layout = contiguous slabs
//    wf[coBlk][ks][row 0..127][kl 0..127], k = ks*128+kl = tap*256+ci
// ---------------------------------------------------------------
__global__ void wconv_kernel(const float* __restrict__ w, _Float16* __restrict__ wf) {
    int i = blockIdx.x * 256 + threadIdx.x;
    if (i >= COUT * KTOT) return;
    int kl    = i & (KSTAGE - 1);
    int row   = (i >> 7) & (COUT_TILE - 1);
    int ks    = (i >> 14) % NSTAGE;
    int coBlk = i / (NSTAGE * COUT_TILE * KSTAGE);
    int co  = coBlk * COUT_TILE + row;
    int k   = ks * KSTAGE + kl;
    int tap = k >> 8;
    int ci  = k & 255;
    float v = w[(co * CIN + ci) * 9 + tap];
    wf[i] = (v >= 0.0f) ? (_Float16)1.0f : (_Float16)(-1.0f);
}

// ---------------------------------------------------------------
// 2) implicit-GEMM binary conv via v_wmma_f32_16x16x32_f16
//    grid: (MTOT/PIX_TILE, COUT/COUT_TILE) ; block 256 (8 waves)
// ---------------------------------------------------------------
__launch_bounds__(256, 2)
__global__ void binconv_wmma_kernel(const float* __restrict__ x,
                                    const _Float16* __restrict__ wf,
                                    _Float16* __restrict__ y,
                                    float* __restrict__ gSum,
                                    float* __restrict__ gSq) {
    __shared__ __align__(64) _Float16 As[COUT_TILE * APITCH]; // 34816 B
    __shared__ __align__(64) _Float16 Bs[PIX_TILE  * BPITCH]; // 17408 B

    const int t    = threadIdx.x;
    const int lane = t & 31;
    const int wv   = t >> 5;
    const int waveCo  = (wv & 3) * 32;       // 0,32,64,96
    const int wavePix = (wv >> 2) * 32;      // 0,32
    const int coBlk  = blockIdx.y;
    const int coBase = coBlk * COUT_TILE;
    const int mBase  = blockIdx.x * PIX_TILE;

    // ---- B-load geometry: one pixel / 32 channels (8 groups of 4) per thread ----
    const int pixL = t >> 2;                 // 0..63
    const int cQ   = (t & 3) * 32;           // 0,32,64,96
    const int m    = mBase + pixL;
    const int ow   = m % OWD;
    const int oh   = (m / OWD) % OH;
    const int nIm  = m / (OWD * OH);

    // ---- A-copy geometry: thread t copies 128 contiguous bytes of the slab ----
    const int rA   = t >> 1;                 // 0..127
    const int hA   = (t & 1) * 64;           // half-row offset in halves

    // ---- fragment lane addressing (contiguous k = 16*hi + s, same order in A and B) ----
    const int rowF = lane & 15;
    const int hi   = lane >> 4;
    const int kbA  = hi * 16;                // A 16x32 f16 lane layout
    const int kbB  = hi * 16;                // B 32x16 f16 lane layout

    v8f c00 = {}, c01 = {}, c10 = {}, c11 = {};

    const _Float16* slab0 = wf + (size_t)coBlk * NSTAGE * COUT_TILE * KSTAGE;

    for (int ks = 0; ks < NSTAGE; ++ks) {
        // ---- stage A: flat 32KB slab -> LDS (async engine if available) ----
        {
            const _Float16* gA = slab0 + (size_t)ks * (COUT_TILE * KSTAGE) + rA * KSTAGE + hA;
            _Float16* lA = As + rA * APITCH + hA;
#ifdef HAVE_ASYNC_LDS
            #pragma unroll
            for (int i = 0; i < 8; ++i) {
                __builtin_amdgcn_global_load_async_to_lds_b128(
                    (gas_v4i_p)(gA + i * 8), (las_v4i_p)(lA + i * 8), 0, 0);
            }
#else
            const uint4* s = (const uint4*)gA;
            uint4* d = (uint4*)lA;
            #pragma unroll
            for (int i = 0; i < 8; ++i) d[i] = s[i];
#endif
        }
        // ---- stage B: 64 pixels x 128 channels, im2col gather f32->f16 ----
        {
            const int tap   = ks >> 1;              // constant within stage
            const int cbase = (ks & 1) * KSTAGE;
            const int kh = tap / 3, kw = tap % 3;
            int ih = oh + kh - 1; ih = ih < 0 ? 0 : (ih > HH - 1 ? HH - 1 : ih);  // replicate H
            const int iwp = ow + kw;                                              // 0..59
            const bool zcol = (iwp == 0) | (iwp == OWD + 1);                      // zero-pad cols
            int iw = iwp - 2; iw = iw < 0 ? 0 : (iw > WW - 1 ? WW - 1 : iw);      // replicate W
            const float* xp = x + (((size_t)nIm * CIN + (cbase + cQ)) * HH + ih) * WW + iw;
            _Float16* bRow = Bs + pixL * BPITCH + cQ;
            #pragma unroll
            for (int g = 0; g < 8; ++g) {
                float f0 = zcol ? 0.0f : xp[(size_t)(g * 4 + 0) * HW];
                float f1 = zcol ? 0.0f : xp[(size_t)(g * 4 + 1) * HW];
                float f2 = zcol ? 0.0f : xp[(size_t)(g * 4 + 2) * HW];
                float f3 = zcol ? 0.0f : xp[(size_t)(g * 4 + 3) * HW];
                v4h hv = { (_Float16)f0, (_Float16)f1, (_Float16)f2, (_Float16)f3 };
                *(v4h*)(bRow + g * 4) = hv;          // ds_store_b64
            }
        }
#ifdef HAVE_ASYNC_LDS
#if __has_builtin(__builtin_amdgcn_s_wait_asynccnt)
        __builtin_amdgcn_s_wait_asynccnt(0);
#else
        asm volatile("s_wait_asynccnt 0" ::: "memory");
#endif
#endif
        __syncthreads();

        // ---- compute: 4 chunks of K=32 -> 16 WMMAs per stage ----
        #pragma unroll
        for (int ch = 0; ch < 4; ++ch) {
            const int ko = ch * 32;
            Frag16 a0, a1, b0, b1;
            const _Float16* aR0 = As + (waveCo + rowF)      * APITCH + ko;
            const _Float16* aR1 = As + (waveCo + rowF + 16) * APITCH + ko;
            const _Float16* bR0 = Bs + (wavePix + rowF)      * BPITCH + ko;
            const _Float16* bR1 = Bs + (wavePix + rowF + 16) * BPITCH + ko;
            a0.h[0] = *(const v8h*)(aR0 + kbA);
            a0.h[1] = *(const v8h*)(aR0 + kbA + 8);
            a1.h[0] = *(const v8h*)(aR1 + kbA);
            a1.h[1] = *(const v8h*)(aR1 + kbA + 8);
            b0.h[0] = *(const v8h*)(bR0 + kbB);
            b0.h[1] = *(const v8h*)(bR0 + kbB + 8);
            b1.h[0] = *(const v8h*)(bR1 + kbB);
            b1.h[1] = *(const v8h*)(bR1 + kbB + 8);

            c00 = __builtin_amdgcn_wmma_f32_16x16x32_f16(false, a0.v, false, b0.v, (short)0, c00, false, false);
            c01 = __builtin_amdgcn_wmma_f32_16x16x32_f16(false, a0.v, false, b1.v, (short)0, c01, false, false);
            c10 = __builtin_amdgcn_wmma_f32_16x16x32_f16(false, a1.v, false, b0.v, (short)0, c10, false, false);
            c11 = __builtin_amdgcn_wmma_f32_16x16x32_f16(false, a1.v, false, b1.v, (short)0, c11, false, false);
        }
        __syncthreads();
    }

    // ---- epilogue: write y (f16, [cout][m]) + per-channel sum / sumsq ----
    // D layout: VGPR r, lane l -> row = r + 8*(l>>4), col = l&15
    const int colLo = lane & 15;

    #pragma unroll
    for (int pair = 0; pair < 2; ++pair) {
        const v8f& ca = pair ? c10 : c00;    // pix cols +0
        const v8f& cb = pair ? c11 : c01;    // pix cols +16
        const int coOff = pair * 16;

        float rs[8], rq[8];
        #pragma unroll
        for (int r = 0; r < 8; ++r) {
            float va = ca[r], vb = cb[r];
            const int coG = coBase + waveCo + coOff + r + 8 * hi;
            const int ma  = mBase + wavePix + colLo;
            y[(size_t)coG * MTOT + ma     ] = (_Float16)va;
            y[(size_t)coG * MTOT + ma + 16] = (_Float16)vb;
            rs[r] = va + vb;
            rq[r] = va * va + vb * vb;
        }
        #pragma unroll
        for (int mask = 1; mask < 16; mask <<= 1) {
            #pragma unroll
            for (int r = 0; r < 8; ++r) {
                rs[r] += __shfl_xor(rs[r], mask, 32);
                rq[r] += __shfl_xor(rq[r], mask, 32);
            }
        }
        if (colLo == 0) {
            #pragma unroll
            for (int r = 0; r < 8; ++r) {
                const int coG = coBase + waveCo + coOff + r + 8 * hi;
                atomicAdd(&gSum[coG], rs[r]);
                atomicAdd(&gSq[coG],  rq[r]);
            }
        }
    }
}

// ---------------------------------------------------------------
// 3) finalize BN: scale = gamma*rsqrt(var+eps), shift = beta - mean*scale
// ---------------------------------------------------------------
__global__ void stats_kernel(const float* __restrict__ gSum, const float* __restrict__ gSq,
                             const float* __restrict__ gamma, const float* __restrict__ beta,
                             float* __restrict__ scale, float* __restrict__ shift) {
    int c = threadIdx.x;
    const float invM = 1.0f / (float)MTOT;
    float mean = gSum[c] * invM;
    float var  = gSq[c] * invM - mean * mean;
    float s    = gamma[c] * rsqrtf(var + 1e-5f);
    scale[c] = s;
    shift[c] = beta[c] - mean * s;
}

// ---------------------------------------------------------------
// 4) binarize: out[n][co][p] = sign(y*scale+shift)   (sign(0)=+1)
// ---------------------------------------------------------------
__global__ void binarize_kernel(const _Float16* __restrict__ y,
                                const float* __restrict__ scale,
                                const float* __restrict__ shift,
                                float* __restrict__ out) {
    const int m  = blockIdx.x * 256 + threadIdx.x;   // 0..MTOT-1
    const int co = blockIdx.y;
    float v = (float)y[(size_t)co * MTOT + m];
    float z = v * scale[co] + shift[co];
    const int n = m / (OH * OWD);
    const int p = m - n * (OH * OWD);
    out[((size_t)n * COUT + co) * (OH * OWD) + p] = (z >= 0.0f) ? 1.0f : -1.0f;
}

// ---------------------------------------------------------------
extern "C" void kernel_launch(void* const* d_in, const int* in_sizes, int n_in,
                              void* d_out, int out_size, void* d_ws, size_t ws_size,
                              hipStream_t stream) {
    const float* x     = (const float*)d_in[0];
    const float* wt    = (const float*)d_in[1];
    const float* gamma = (const float*)d_in[2];
    const float* beta  = (const float*)d_in[3];
    float* out = (float*)d_out;

    char* ws = (char*)d_ws;
    _Float16* wf16 = (_Float16*)ws;                       // 1,179,648 B
    size_t off = (size_t)COUT * KTOT * sizeof(_Float16);
    float* gSum  = (float*)(ws + off);  off += 1024;
    float* gSq   = (float*)(ws + off);  off += 1024;
    float* scale = (float*)(ws + off);  off += 1024;
    float* shift = (float*)(ws + off);  off += 1024;
    _Float16* y  = (_Float16*)(ws + off);                 // 53,215,232 B

    zero_stats_kernel<<<1, 512, 0, stream>>>(gSum, gSq);

    wconv_kernel<<<(COUT * KTOT + 255) / 256, 256, 0, stream>>>(wt, wf16);

    dim3 cgrid(MTOT / PIX_TILE, COUT / COUT_TILE);        // (1624, 2)
    binconv_wmma_kernel<<<cgrid, 256, 0, stream>>>(x, wf16, y, gSum, gSq);

    stats_kernel<<<1, COUT, 0, stream>>>(gSum, gSq, gamma, beta, scale, shift);

    dim3 bgrid(MTOT / 256, COUT);                         // (406, 256)
    binarize_kernel<<<bgrid, 256, 0, stream>>>(y, scale, shift, out);
}
